// RRN_20229295964406
// MI455X (gfx1250) — matrix operand order repaired and checked by reading, not compiled, not run on hardware
//
#include <hip/hip_runtime.h>
#include <hip/hip_bf16.h>
#include <stdint.h>

// ---------------------------------------------------------------------------
// Problem constants (from the reference): B=128, T=512, D=256, H=1024, C=512
// ---------------------------------------------------------------------------
#define B_  128
#define T_  512
#define D_  256
#define H_  1024
#define C_  512

typedef __bf16 bf16;
typedef __attribute__((ext_vector_type(16))) __bf16   v16bf;
typedef __attribute__((ext_vector_type(8)))  __bf16   v8bf;
typedef __attribute__((ext_vector_type(8)))  float    v8f;
typedef __attribute__((ext_vector_type(4)))  unsigned u32x4;   // trivial 128-bit chunk

union V16U { v16bf v; u32x4 u[2]; };   // 16 bf16 = 32B (WMMA A/B operand per lane)
union V8U  { v8bf  v; u32x4 u;    };   // 8  bf16 = 16B

// ---------------------------------------------------------------------------
// Workspace layout (bytes). Needs ~296 MB total.
//   Bph : [Wih|Wfh] packed bf16, B-operand layout [nt=128][kt=32][lane=32][16]
//   Bpx : [Wix|Wfx] packed bf16,               [nt=128][kt= 8][lane=32][16]
//   Bpp : Wph packed bf16,                     [nt= 32][kt=32][lane=32][16]
//   h0/h1 : h state, row-major bf16 [128][1024], double buffered
//   bar : grid barrier counters
//   xA  : x pre-packed bf16, A-operand layout  [mtile=4096][kt=8][lane=32][16]
//   gxp : precomputed x-gates + bias, bf16, accumulator layout
//         [mtile=4096][nt=128][lane=32][8]   (mtile = t*8 + batch-tile)
// ---------------------------------------------------------------------------
#define BPH_OFF  ((size_t)0)           // 4,194,304 B
#define BPX_OFF  ((size_t)4194304)     // 1,048,576 B
#define BPP_OFF  ((size_t)5242880)     // 1,048,576 B
#define H0_OFF   ((size_t)6291456)     //   262,144 B
#define H1_OFF   ((size_t)6553600)     //   262,144 B
#define BAR_OFF  ((size_t)6815744)     //       256 B
#define XA_OFF   ((size_t)8388608)     // 33,554,432 B
#define GXP_OFF  ((size_t)41943040)    // 268,435,456 B

// ---------------------------------------------------------------------------
// Weight pack kernels: fp32 -> bf16, swizzled into the WMMA B-operand layout:
//   lane c holds column n = nt*16 + (c&15); its 16 values are two 8-element
//   K groups at kt*32 + (c>>4)*8 and kt*32 + 16 + (c>>4)*8.
// ---------------------------------------------------------------------------
__global__ void pack_wh_kernel(const float* __restrict__ Wih,
                               const float* __restrict__ Wfh,
                               bf16* __restrict__ out) {
  int id = blockIdx.x * 256 + threadIdx.x;          // 2,097,152 elements
  int j    =  id        & 15;
  int lane = (id >> 4)  & 31;
  int kt   = (id >> 9)  & 31;
  int nt   =  id >> 14;                             // 0..127
  int n = nt * 16 + (lane & 15);                    // 0..2047  ([Wih | Wfh])
  int k = kt * 32 + ((j >> 3) << 4) + ((lane >> 4) << 3) + (j & 7);  // 0..1023
  float v = (n < H_) ? Wih[(size_t)k * H_ + n] : Wfh[(size_t)k * H_ + (n - H_)];
  out[id] = (bf16)v;
}

__global__ void pack_wx_kernel(const float* __restrict__ Wix,
                               const float* __restrict__ Wfx,
                               bf16* __restrict__ out) {
  int id = blockIdx.x * 256 + threadIdx.x;          // 524,288 elements
  int j    =  id        & 15;
  int lane = (id >> 4)  & 31;
  int kt   = (id >> 9)  & 7;
  int nt   =  id >> 12;                             // 0..127
  int n = nt * 16 + (lane & 15);
  int k = kt * 32 + ((j >> 3) << 4) + ((lane >> 4) << 3) + (j & 7);  // 0..255
  float v = (n < H_) ? Wix[(size_t)k * H_ + n] : Wfx[(size_t)k * H_ + (n - H_)];
  out[id] = (bf16)v;
}

__global__ void pack_wp_kernel(const float* __restrict__ Wph,
                               bf16* __restrict__ out) {
  int id = blockIdx.x * 256 + threadIdx.x;          // 524,288 elements
  int j    =  id        & 15;
  int lane = (id >> 4)  & 31;
  int kt   = (id >> 9)  & 31;
  int nt   =  id >> 14;                             // 0..31
  int n = nt * 16 + (lane & 15);                    // 0..511
  int k = kt * 32 + ((j >> 3) << 4) + ((lane >> 4) << 3) + (j & 7);  // 0..1023
  out[id] = (bf16)Wph[(size_t)k * C_ + n];
}

// ---------------------------------------------------------------------------
// Pack x (fp32, [B][T][D]) -> bf16 in the WMMA A-operand layout:
//   xA[mtile][kt][lane][16], lane l holds row m = mtile*16 + (l&15)
//   (m = t*128 + b), two 8-element K groups at kt*32 + khi and kt*32+16+khi,
//   khi = (l>>4)*8.  One thread converts one contiguous 8-float group ->
//   one 16 B store.
// ---------------------------------------------------------------------------
__global__ void pack_x_kernel(const float* __restrict__ x, bf16* __restrict__ xA) {
  int id = blockIdx.x * 256 + threadIdx.x;          // 2,097,152 threads
  int half  =  id       & 1;
  int lane  = (id >> 1) & 31;
  int kt    = (id >> 6) & 7;
  int mtile =  id >> 9;                             // 0..4095
  int row = mtile * 16 + (lane & 15);               // m = t*128 + b
  int t = row >> 7, b = row & 127;
  int khi = (lane >> 4) << 3;
  int k = kt * 32 + (half ? 16 + khi : khi);
  const float* p = x + ((size_t)b * T_ + t) * D_ + k;   // 8 contiguous floats
  V8U o;
#pragma unroll
  for (int j = 0; j < 8; ++j) o.v[j] = (bf16)p[j];
  size_t eoff = (((size_t)mtile * 8 + kt) * 32 + lane) * 16 + (size_t)half * 8;
  *(u32x4*)(xA + eoff) = o.u;
}

// Zero h0 (initial hidden state) and the grid-barrier counters every launch
// (determinism requirement).
__global__ void init_kernel(unsigned* __restrict__ h0u, unsigned* __restrict__ bar) {
  int id = blockIdx.x * 256 + threadIdx.x;          // 65,536 threads
  h0u[id] = 0u;                                     // 262,144 B of h0
  if (id < 64) bar[id] = 0u;
}

// ---------------------------------------------------------------------------
// Phase 1: gate_x = x @ [Wix|Wfx] + [bi|bf] for ALL timesteps.
// GEMM [65536, 256] x [256, 2048].  Block = 8 waves (8 batch tiles) sharing
// one group of 4 n-tiles whose B slices (64 KB) are LDS-staged once.  Each
// wave keeps 4 independent accumulators -> 4 WMMAs per A load, no fp32->bf16
// conversion in the loop (x pre-packed to bf16 A layout).
// ---------------------------------------------------------------------------
__global__ void __launch_bounds__(256) gatex_kernel(
    const bf16* __restrict__ xA,    // [4096][8][32][16]
    const bf16* __restrict__ Bpx,   // [128][8][32][16]
    const float* __restrict__ bi,
    const float* __restrict__ bfv,
    bf16*        __restrict__ gxp) {
  __shared__ u32x4 ldsB[4][1024];   // 4 n-tiles x 16 KB = 64 KB
  const int tid  = threadIdx.x;
  const int lane = tid & 31;
  const int wid  = tid >> 5;
  const int ntg  = blockIdx.x & 31;           // n-tile group: tiles ntg*4 .. +3
  const int mg   = blockIdx.x >> 5;           // 0..511
  const int mtile = mg * 8 + wid;             // 0..4095

  // Stage 4 consecutive B slices (contiguous in Bpx) into LDS.
  {
    const u32x4* src = (const u32x4*)Bpx + (size_t)ntg * 4096;
    u32x4* dst = &ldsB[0][0];
    for (int i = tid; i < 4096; i += 256) dst[i] = src[i];
  }
  __syncthreads();

  v8f acc0 = {}, acc1 = {}, acc2 = {}, acc3 = {};
  const u32x4* ax = (const u32x4*)xA + (size_t)mtile * 8 * 64;  // per kt: 64 u32x4

#pragma unroll
  for (int kt = 0; kt < 8; ++kt) {
    V16U a;
    a.u[0] = ax[kt * 64 + lane * 2];
    a.u[1] = ax[kt * 64 + lane * 2 + 1];
    const int bidx = kt * 64 + lane * 2;
    V16U b0, b1, b2, b3;
    b0.u[0] = ldsB[0][bidx]; b0.u[1] = ldsB[0][bidx + 1];
    b1.u[0] = ldsB[1][bidx]; b1.u[1] = ldsB[1][bidx + 1];
    b2.u[0] = ldsB[2][bidx]; b2.u[1] = ldsB[2][bidx + 1];
    b3.u[0] = ldsB[3][bidx]; b3.u[1] = ldsB[3][bidx + 1];
    acc0 = __builtin_amdgcn_wmma_f32_16x16x32_bf16(false, a.v, false, b0.v, (short)0, acc0, false, false);
    acc1 = __builtin_amdgcn_wmma_f32_16x16x32_bf16(false, a.v, false, b1.v, (short)0, acc1, false, false);
    acc2 = __builtin_amdgcn_wmma_f32_16x16x32_bf16(false, a.v, false, b2.v, (short)0, acc2, false, false);
    acc3 = __builtin_amdgcn_wmma_f32_16x16x32_bf16(false, a.v, false, b3.v, (short)0, acc3, false, false);
  }

  v8f accs[4] = {acc0, acc1, acc2, acc3};
#pragma unroll
  for (int i = 0; i < 4; ++i) {
    const int nt   = ntg * 4 + i;
    const int ncol = nt * 16 + (lane & 15);
    const float bias = (ncol < H_) ? bi[ncol] : bfv[ncol - H_];
    V8U o;
#pragma unroll
    for (int r = 0; r < 8; ++r) o.v[r] = (bf16)(accs[i][r] + bias);
    *((u32x4*)gxp + ((size_t)mtile * 128 + nt) * 32 + lane) = o.u;
  }
}

// ---------------------------------------------------------------------------
// Grid-wide sense barrier (monotone generation counter, reset each launch).
// ---------------------------------------------------------------------------
__device__ __forceinline__ void grid_sync(unsigned* bar, unsigned nblk, unsigned target) {
  __syncthreads();
  if (threadIdx.x == 0) {
    __threadfence();
    unsigned prev = atomicAdd(&bar[0], 1u);
    if (prev == nblk - 1u) {
      atomicExch(&bar[0], 0u);
      __threadfence();
      atomicAdd(&bar[1], 1u);                       // release generation
    } else {
      while (__hip_atomic_load(&bar[1], __ATOMIC_ACQUIRE, __HIP_MEMORY_SCOPE_AGENT) < target)
        __builtin_amdgcn_s_sleep(2);
    }
  }
  __syncthreads();
}

// ---------------------------------------------------------------------------
// Phase 2: persistent recurrent kernel. 64 blocks x 8 waves = 512 waves.
// Block ht owns h-feature tile ht; wave mt owns batch tile mt. Each wave
// computes BOTH the i-gate and f-gate 16x16 tiles for (mt, ht), so the cell
// state c is 8 persistent VGPRs. Recurrent weights live in LDS for all 512
// steps; h double-buffers in L2.  64 WMMA per wave per step.
// ---------------------------------------------------------------------------
__global__ void __launch_bounds__(256, 1) rnn_step_kernel(
    const bf16* __restrict__ Bph,   // [128][32][32][16] bf16
    const bf16* __restrict__ gxp,   // [4096][128][32][8] bf16
    bf16* __restrict__ h0,
    bf16* __restrict__ h1,
    unsigned* __restrict__ bar) {
  __shared__ u32x4 ldsB[2][2048];   // per gate: 32 kt x 32 lane x 32B = 32 KB

  const int tid  = threadIdx.x;
  const int lane = tid & 31;
  const int mt   = tid >> 5;        // 0..7  (batch tile)
  const int ht   = blockIdx.x;      // 0..63 (h-feature tile)

  // Stage recurrent weight tiles once (constant over time).
  {
    const u32x4* srcI = (const u32x4*)Bph + (size_t)ht * 2048;          // i-gate cols
    const u32x4* srcF = (const u32x4*)Bph + (size_t)(64 + ht) * 2048;   // f-gate cols
    for (int i = tid; i < 2048; i += 256) { ldsB[0][i] = srcI[i]; ldsB[1][i] = srcF[i]; }
  }
  __syncthreads();

  const int khi = (lane >> 4) << 3;
  float cst[8];
#pragma unroll
  for (int r = 0; r < 8; ++r) cst[r] = 0.f;

  bf16* hcur = h0;                  // zero-initialized by init_kernel
  bf16* hnxt = h1;

  for (int t = 0; t < T_; ++t) {
    v8f ai = {}, af = {};
    const bf16* hrow = hcur + (size_t)(mt * 16 + (lane & 15)) * H_;
#pragma unroll
    for (int kt = 0; kt < 32; ++kt) {
      V16U a, bi_t, bf_t;
      a.u[0] = *(const u32x4*)(hrow + kt * 32 + khi);
      a.u[1] = *(const u32x4*)(hrow + kt * 32 + 16 + khi);
      const int bidx = (kt * 32 + lane) * 2;
      bi_t.u[0] = ldsB[0][bidx]; bi_t.u[1] = ldsB[0][bidx + 1];
      bf_t.u[0] = ldsB[1][bidx]; bf_t.u[1] = ldsB[1][bidx + 1];
      ai = __builtin_amdgcn_wmma_f32_16x16x32_bf16(false, a.v, false, bi_t.v,
                                                   (short)0, ai, false, false);
      af = __builtin_amdgcn_wmma_f32_16x16x32_bf16(false, a.v, false, bf_t.v,
                                                   (short)0, af, false, false);
    }

    // Precomputed x-projection + bias (accumulator layout, one b128 per gate).
    V8U gi, gf;
    const size_t gbase = (size_t)(t * 8 + mt) * 128;
    gi.u = *((const u32x4*)gxp + (gbase + ht) * 32 + lane);
    gf.u = *((const u32x4*)gxp + (gbase + 64 + ht) * 32 + lane);

    const int mrow = mt * 16 + ((lane >> 4) << 3);
    const int ncol = ht * 16 + (lane & 15);
#pragma unroll
    for (int r = 0; r < 8; ++r) {
      float iv = ai[r] + (float)gi.v[r]; iv = iv > 0.f ? iv : 0.f;
      float fv = af[r] + (float)gf.v[r]; fv = fv > 0.f ? fv : 0.f;
      cst[r] += iv - fv;                              // c = c + i - f
      float hv = cst[r] > 0.f ? cst[r] : 0.f;         // h = relu(c)
      hnxt[(size_t)(mrow + r) * H_ + ncol] = (bf16)hv;
    }

    { bf16* tmp = hnxt; hnxt = hcur; hcur = tmp; }
    grid_sync(bar, 64u, (unsigned)(t + 1));
  }
  // T even -> final h resides in h0.
}

// ---------------------------------------------------------------------------
// Phase 3: out = h_final @ Wph + bp.  [128,1024]x[1024,512].
// 32 blocks (one N-tile each, weights staged in LDS) x 8 waves (batch tiles).
// ---------------------------------------------------------------------------
__global__ void __launch_bounds__(256) proj_kernel(
    const bf16* __restrict__ hfin,
    const bf16* __restrict__ Bpp,
    const float* __restrict__ bp,
    float* __restrict__ out) {
  __shared__ u32x4 ldsB[2048];
  const int tid = threadIdx.x, lane = tid & 31, mt = tid >> 5;
  const int nt = blockIdx.x;                          // 0..31
  const u32x4* src = (const u32x4*)Bpp + (size_t)nt * 2048;
  for (int i = tid; i < 2048; i += 256) ldsB[i] = src[i];
  __syncthreads();

  const int khi = (lane >> 4) << 3;
  const bf16* hrow = hfin + (size_t)(mt * 16 + (lane & 15)) * H_;
  v8f acc = {};
#pragma unroll
  for (int kt = 0; kt < 32; ++kt) {
    V16U a, bb;
    a.u[0] = *(const u32x4*)(hrow + kt * 32 + khi);
    a.u[1] = *(const u32x4*)(hrow + kt * 32 + 16 + khi);
    const int bidx = (kt * 32 + lane) * 2;
    bb.u[0] = ldsB[bidx]; bb.u[1] = ldsB[bidx + 1];
    acc = __builtin_amdgcn_wmma_f32_16x16x32_bf16(false, a.v, false, bb.v,
                                                  (short)0, acc, false, false);
  }
  const int ncol = nt * 16 + (lane & 15);
  const float bias = bp[ncol];
  const int mrow = mt * 16 + ((lane >> 4) << 3);
#pragma unroll
  for (int r = 0; r < 8; ++r) out[(size_t)(mrow + r) * C_ + ncol] = acc[r] + bias;
}

// ---------------------------------------------------------------------------
// Launcher. Input order: x, Wix, Wih, bi, Wfx, Wfh, bf, Wph, bp.
// Requires ws_size >= ~296 MB (gate_x precompute dominates).
// ---------------------------------------------------------------------------
extern "C" void kernel_launch(void* const* d_in, const int* in_sizes, int n_in,
                              void* d_out, int out_size, void* d_ws, size_t ws_size,
                              hipStream_t stream) {
  (void)in_sizes; (void)n_in; (void)out_size; (void)ws_size;
  const float* x   = (const float*)d_in[0];
  const float* Wix = (const float*)d_in[1];
  const float* Wih = (const float*)d_in[2];
  const float* bi  = (const float*)d_in[3];
  const float* Wfx = (const float*)d_in[4];
  const float* Wfh = (const float*)d_in[5];
  const float* bfv = (const float*)d_in[6];
  const float* Wph = (const float*)d_in[7];
  const float* bp  = (const float*)d_in[8];

  char* ws = (char*)d_ws;
  bf16*     Bph = (bf16*)(ws + BPH_OFF);
  bf16*     Bpx = (bf16*)(ws + BPX_OFF);
  bf16*     Bpp = (bf16*)(ws + BPP_OFF);
  bf16*     h0  = (bf16*)(ws + H0_OFF);
  bf16*     h1  = (bf16*)(ws + H1_OFF);
  unsigned* bar = (unsigned*)(ws + BAR_OFF);
  bf16*     xA  = (bf16*)(ws + XA_OFF);
  bf16*     gxp = (bf16*)(ws + GXP_OFF);

  hipLaunchKernelGGL(pack_wh_kernel,  dim3(8192),  dim3(256), 0, stream, Wih, Wfh, Bph);
  hipLaunchKernelGGL(pack_wx_kernel,  dim3(2048),  dim3(256), 0, stream, Wix, Wfx, Bpx);
  hipLaunchKernelGGL(pack_wp_kernel,  dim3(2048),  dim3(256), 0, stream, Wph, Bpp);
  hipLaunchKernelGGL(pack_x_kernel,   dim3(8192),  dim3(256), 0, stream, x, xA);
  hipLaunchKernelGGL(init_kernel,     dim3(256),   dim3(256), 0, stream, (unsigned*)h0, bar);
  hipLaunchKernelGGL(gatex_kernel,    dim3(16384), dim3(256), 0, stream, xA, Bpx, bi, bfv, gxp);
  hipLaunchKernelGGL(rnn_step_kernel, dim3(64),    dim3(256), 0, stream, Bph, gxp, h0, h1, bar);
  hipLaunchKernelGGL(proj_kernel,     dim3(32),    dim3(256), 0, stream, h0, Bpp, bp, (float*)d_out);
}
